// Block_29575144800619
// MI455X (gfx1250) — matrix operand order repaired and checked by reading, not compiled, not run on hardware
//
#include <hip/hip_runtime.h>
#include <hip/hip_bf16.h>
#include <math.h>

// Problem constants (match reference)
#define B_   4
#define T_   2048
#define E_   1024
#define H_   16
#define HS_  64
#define DFF_ 4096
#define M_   (B_ * T_)
#define EPS_ 1e-5f

typedef __bf16 bf16_t;
typedef bf16_t v16bf __attribute__((ext_vector_type(16)));
typedef float  v8f   __attribute__((ext_vector_type(8)));

// float -> bf16 round-to-nearest-even via bit math (no __bf16 arithmetic needed)
__device__ __forceinline__ bf16_t f2bf(float f) {
    union { float f; unsigned u; } a; a.f = f;
    unsigned r = a.u + 0x7FFFu + ((a.u >> 16) & 1u);
    unsigned short h = (unsigned short)(r >> 16);
    bf16_t out; __builtin_memcpy(&out, &h, 2);
    return out;
}

// Async 16B copy global -> LDS (GLOBAL_LOAD_ASYNC_TO_LDS_B128, ASYNCcnt).
// Per-lane: LDS[lds_byte_off] = MEM[gptr], 16 bytes.
__device__ __forceinline__ void async_copy16(unsigned lds_byte_off, const bf16_t* gptr) {
    asm volatile("global_load_async_to_lds_b128 %0, %1, off"
                 :: "v"(lds_byte_off), "v"(gptr) : "memory");
}
__device__ __forceinline__ void wait_async0() {
    asm volatile("s_wait_asynccnt 0x0" ::: "memory");
}

// ---------------------------------------------------------------------------
// WMMA fragment loaders (CDNA5 wave32 layouts, CDNA5 ISA §7.12.2)
// A (16x32 bf16, row-major source, ld = lda):
//   lane m = lane&15, half = lane>>4
//   elements 0..7  -> K = k0 + half*8  + e        (16 contiguous bytes)
//   elements 8..15 -> K = k0 + 16 + half*8 + e-8  (16 contiguous bytes)
__device__ __forceinline__ v16bf load_a_frag(const bf16_t* A, int lda, int row0, int k0) {
    int lane = threadIdx.x & 31;
    int m = lane & 15, hf = (lane >> 4) & 1;
    const bf16_t* p = A + (size_t)(row0 + m) * lda + k0 + hf * 8;
    v16bf a;
    ((uint4*)&a)[0] = *(const uint4*)(p);
    ((uint4*)&a)[1] = *(const uint4*)(p + 16);
    return a;
}

// B (32x16 bf16) from a TRANSPOSED source Bt[n][k] (ld = ldb):
//   lane n = lane&15, half = lane>>4 ; elements 0..7 -> K = k0 + half*8 + e, 8..15 -> K = k0 + 16 + half*8 + e-8
__device__ __forceinline__ v16bf load_bt_frag(const bf16_t* Bt, int ldb, int k0, int col0) {
    int lane = threadIdx.x & 31;
    int n = lane & 15, hf = (lane >> 4) & 1;
    const bf16_t* p = Bt + (size_t)(col0 + n) * ldb + k0 + hf * 8;
    v16bf b;
    ((uint4*)&b)[0] = *(const uint4*)(p);
    ((uint4*)&b)[1] = *(const uint4*)(p + 16);
    return b;
}

// ---------------------------------------------------------------------------
// Weight convert + transpose: Wt[n][k] = bf16(W[k][n]); K,N multiples of 32
__global__ __launch_bounds__(256) void conv_transpose_bf16(
    const float* __restrict__ W, bf16_t* __restrict__ Wt, int Kdim, int Ndim) {
    __shared__ float tile[32][33];
    int k0 = blockIdx.x * 32, n0 = blockIdx.y * 32;
    int tx = threadIdx.x & 31, ty = threadIdx.x >> 5;
#pragma unroll
    for (int i = 0; i < 4; ++i)
        tile[ty + 8 * i][tx] = W[(size_t)(k0 + ty + 8 * i) * Ndim + n0 + tx];
    __syncthreads();
#pragma unroll
    for (int i = 0; i < 4; ++i)
        Wt[(size_t)(n0 + ty + 8 * i) * Kdim + k0 + tx] = f2bf(tile[tx][ty + 8 * i]);
}

// ---------------------------------------------------------------------------
// LayerNorm over E=1024, one row per block (8 waves), output bf16
__global__ __launch_bounds__(256) void layernorm_bf16(
    const float* __restrict__ x, const float* __restrict__ g,
    const float* __restrict__ bb, bf16_t* __restrict__ out) {
    int row = blockIdx.x;
    const float* xr = x + (size_t)row * E_;
    float s = 0.f, sq = 0.f;
    for (int j = threadIdx.x; j < E_; j += 256) { float v = xr[j]; s += v; sq += v * v; }
#pragma unroll
    for (int m = 16; m >= 1; m >>= 1) {
        s  += __shfl_xor(s,  m, 32);
        sq += __shfl_xor(sq, m, 32);
    }
    __shared__ float s1[8], s2[8];
    int wave = threadIdx.x >> 5, lane = threadIdx.x & 31;
    if (lane == 0) { s1[wave] = s; s2[wave] = sq; }
    __syncthreads();
    float ts = 0.f, tq = 0.f;
#pragma unroll
    for (int i = 0; i < 8; ++i) { ts += s1[i]; tq += s2[i]; }
    float mu  = ts * (1.0f / E_);
    float var = tq * (1.0f / E_) - mu * mu;
    float inv = rsqrtf(var + EPS_);
    for (int j = threadIdx.x; j < E_; j += 256)
        out[(size_t)row * E_ + j] = f2bf((xr[j] - mu) * inv * g[j] + bb[j]);
}

// ---------------------------------------------------------------------------
// WMMA GEMM, async double-buffered LDS staging + 2x2 register blocking:
//   block tile 64(M) x 128(N), K stepped by 32
//   A chunk 64x32, B chunk 128x32 staged via GLOBAL_LOAD_ASYNC_TO_LDS_B128
//   LDS rows padded to 40 halves (80B) -> conflict-free ds_load_b128 frags
//   wave tile 32x32 (2 A-frags x 2 B-frags -> 4 WMMA / K-step)
// OUT_MODE: 0 f32 row-major, 1 bf16 row-major,
//           2 bf16 per-head [B,H,T,HS], 3 bf16 per-head transposed [B,H,HS,T]
#define LDSTRIDE 40                      // halves per staged row (32 + 8 pad)
#define A_HALVES (64 * LDSTRIDE)         // 2560 halves
#define BUF_HALVES (A_HALVES + 128 * LDSTRIDE)  // 7680 halves = 15360 B

template <int OUT_MODE, bool RELU, bool RES>
__global__ __launch_bounds__(256) void gemm_bf16(
    const bf16_t* __restrict__ A, const bf16_t* __restrict__ Bt,
    const float* __restrict__ bias, const float* __restrict__ resid,
    void* __restrict__ outp, int Mdim, int Ndim, int Kdim) {
    __shared__ __align__(16) bf16_t smem[2 * BUF_HALVES];
    int tid  = threadIdx.x;
    int wave = tid >> 5;
    int lane = tid & 31;
    int n16 = lane & 15, hf = (lane >> 4) & 1;
    int wm = wave >> 2;                       // 0..1 (M direction)
    int wn = wave & 3;                        // 0..3 (N direction)
    int rowA0 = blockIdx.x * 64;              // block M origin
    int colB0 = blockIdx.y * 128;             // block N origin
    int row0 = rowA0 + wm * 32;               // wave rows row0 .. row0+31
    int col0 = colB0 + wn * 32;               // wave cols col0 .. col0+31

    unsigned smembase = (unsigned)(uintptr_t)smem;

    // Stage K-chunk k0 into buffer `buf` (3 async b128 copies per thread)
    auto stage = [&](int buf, int k0) {
        unsigned lb = smembase + (unsigned)buf * (BUF_HALVES * 2);
        {   // A: 64 rows x 4 chunks of 16B; chunk id = tid
            int row = tid >> 2, c = tid & 3;
            async_copy16(lb + (unsigned)(row * LDSTRIDE + c * 8) * 2,
                         A + (size_t)(rowA0 + row) * Kdim + k0 + c * 8);
        }
#pragma unroll
        for (int i = 0; i < 2; ++i) {  // B: 128 rows x 4 chunks; ids tid, tid+256
            int id = tid + i * 256;
            int row = id >> 2, c = id & 3;
            async_copy16(lb + (unsigned)(A_HALVES + row * LDSTRIDE + c * 8) * 2,
                         Bt + (size_t)(colB0 + row) * Kdim + k0 + c * 8);
        }
    };

    v8f acc00 = {0.f,0.f,0.f,0.f,0.f,0.f,0.f,0.f};
    v8f acc01 = acc00, acc10 = acc00, acc11 = acc00;

    int nK = Kdim >> 5;
    stage(0, 0);
    for (int kt = 0; kt < nK; ++kt) {
        int cur = kt & 1;
        wait_async0();        // my async copies for buffer `cur` done
        __syncthreads();      // everyone's done (also drains my DS reads of nxt)
        if (kt + 1 < nK) stage(cur ^ 1, (kt + 1) * 32);

        const bf16_t* As = smem + (size_t)cur * BUF_HALVES;
        const bf16_t* Bs = As + A_HALVES;
        v16bf a0 = load_a_frag(As, LDSTRIDE, wm * 32,      0);
        v16bf a1 = load_a_frag(As, LDSTRIDE, wm * 32 + 16, 0);
        v16bf b0 = load_bt_frag(Bs, LDSTRIDE, 0, wn * 32);
        v16bf b1 = load_bt_frag(Bs, LDSTRIDE, 0, wn * 32 + 16);
        acc00 = __builtin_amdgcn_wmma_f32_16x16x32_bf16(false, a0, false, b0,
                                                        (short)0, acc00, false, false);
        acc01 = __builtin_amdgcn_wmma_f32_16x16x32_bf16(false, a0, false, b1,
                                                        (short)0, acc01, false, false);
        acc10 = __builtin_amdgcn_wmma_f32_16x16x32_bf16(false, a1, false, b0,
                                                        (short)0, acc10, false, false);
        acc11 = __builtin_amdgcn_wmma_f32_16x16x32_bf16(false, a1, false, b1,
                                                        (short)0, acc11, false, false);
    }

    v8f accs[2][2] = {{acc00, acc01}, {acc10, acc11}};
#pragma unroll
    for (int mi = 0; mi < 2; ++mi) {
#pragma unroll
        for (int ni = 0; ni < 2; ++ni) {
            int col = col0 + ni * 16 + n16;
            float bs = bias ? bias[col] : 0.f;
#pragma unroll
            for (int r = 0; r < 8; ++r) {
                int row = row0 + mi * 16 + r + hf * 8;
                float v = accs[mi][ni][r] + bs;
                if (RELU) v = fmaxf(v, 0.f);
                if (RES)  v += resid[(size_t)row * Ndim + col];
                if (OUT_MODE == 0) {
                    ((float*)outp)[(size_t)row * Ndim + col] = v;
                } else if (OUT_MODE == 1) {
                    ((bf16_t*)outp)[(size_t)row * Ndim + col] = f2bf(v);
                } else if (OUT_MODE == 2) {
                    int bi = row / T_, t = row % T_;
                    int hh = col / HS_, d = col % HS_;
                    ((bf16_t*)outp)[(((size_t)(bi * H_ + hh) * T_) + t) * HS_ + d] = f2bf(v);
                } else {
                    int bi = row / T_, t = row % T_;
                    int hh = col / HS_, d = col % HS_;
                    ((bf16_t*)outp)[(((size_t)(bi * H_ + hh) * HS_) + d) * T_ + t] = f2bf(v);
                }
            }
        }
    }
}

// ---------------------------------------------------------------------------
// Flash attention (causal). One wave handles one 16-row q-tile; KV in chunks
// of 32 keys; online softmax in registers; P re-laid out C->A via 1KB LDS.
// Q,K: [B,H,T,HS] bf16. Vt: [B,H,HS,T] bf16. Out: [B,T,E] bf16 row-major.
__global__ __launch_bounds__(256) void flash_attn(
    const bf16_t* __restrict__ Q, const bf16_t* __restrict__ K,
    const bf16_t* __restrict__ Vt, bf16_t* __restrict__ O) {
    __shared__ __align__(16) bf16_t Pbuf[8][16 * 32];
    int wave = threadIdx.x >> 5;
    int lane = threadIdx.x & 31;
    int n16 = lane & 15, hf = (lane >> 4) & 1;
    int bh = blockIdx.y;               // b*H + h
    int b  = bh / H_, h = bh % H_;
    int q0 = (blockIdx.x * 8 + wave) * 16;

    const bf16_t* Qh = Q  + (size_t)bh * T_ * HS_;
    const bf16_t* Kh = K  + (size_t)bh * T_ * HS_;
    const bf16_t* Vh = Vt + (size_t)bh * HS_ * T_;
    bf16_t* Pw = &Pbuf[wave][0];

    v16bf aQ0 = load_a_frag(Qh, HS_, q0, 0);
    v16bf aQ1 = load_a_frag(Qh, HS_, q0, 32);

    v8f accO[4];
#pragma unroll
    for (int t2 = 0; t2 < 4; ++t2) accO[t2] = (v8f){0.f,0.f,0.f,0.f,0.f,0.f,0.f,0.f};
    float mrow[8], lrow[8];
#pragma unroll
    for (int r = 0; r < 8; ++r) { mrow[r] = -INFINITY; lrow[r] = 0.f; }

    const float scale = 0.125f;  // HS^-0.5, HS=64

    for (int kv0 = 0; kv0 < q0 + 16; kv0 += 32) {
        // S chunk = Q(16x64) x K^T(64x32): two 16-col tiles, K-dim split 32+32
        v8f s0 = {0.f,0.f,0.f,0.f,0.f,0.f,0.f,0.f};
        v8f s1 = {0.f,0.f,0.f,0.f,0.f,0.f,0.f,0.f};
        s0 = __builtin_amdgcn_wmma_f32_16x16x32_bf16(false, aQ0, false,
                 load_bt_frag(Kh, HS_, 0,  kv0),      (short)0, s0, false, false);
        s0 = __builtin_amdgcn_wmma_f32_16x16x32_bf16(false, aQ1, false,
                 load_bt_frag(Kh, HS_, 32, kv0),      (short)0, s0, false, false);
        s1 = __builtin_amdgcn_wmma_f32_16x16x32_bf16(false, aQ0, false,
                 load_bt_frag(Kh, HS_, 0,  kv0 + 16), (short)0, s1, false, false);
        s1 = __builtin_amdgcn_wmma_f32_16x16x32_bf16(false, aQ1, false,
                 load_bt_frag(Kh, HS_, 32, kv0 + 16), (short)0, s1, false, false);

        int c0 = kv0 + n16, c1 = kv0 + 16 + n16;
#pragma unroll
        for (int r = 0; r < 8; ++r) {
            int rowg = q0 + r + hf * 8;
            float a0 = s0[r] * scale; if (c0 > rowg) a0 = -INFINITY;
            float a1 = s1[r] * scale; if (c1 > rowg) a1 = -INFINITY;
            // row-wise max across the 16 lanes of this half-wave
            float mx = fmaxf(a0, a1);
#pragma unroll
            for (int msk = 8; msk >= 1; msk >>= 1) mx = fmaxf(mx, __shfl_xor(mx, msk, 32));
            float mnew = fmaxf(mrow[r], mx);
            float p0 = __expf(a0 - mnew), p1 = __expf(a1 - mnew);
            float rs = p0 + p1;
#pragma unroll
            for (int msk = 8; msk >= 1; msk >>= 1) rs += __shfl_xor(rs, msk, 32);
            float corr = __expf(mrow[r] - mnew);
            mrow[r] = mnew;
            lrow[r] = lrow[r] * corr + rs;
#pragma unroll
            for (int t2 = 0; t2 < 4; ++t2) accO[t2][r] *= corr;
            int prow = (r + hf * 8) * 32;
            Pw[prow + n16]      = f2bf(p0);
            Pw[prow + 16 + n16] = f2bf(p1);
        }
        // Intra-wave LDS RAW fence: DS ops are in-order per wave; wait stores.
        asm volatile("s_wait_dscnt 0" ::: "memory");
        v16bf aP = load_a_frag(Pw, 32, 0, 0);  // P as A-fragment (16x32)
#pragma unroll
        for (int t2 = 0; t2 < 4; ++t2)
            accO[t2] = __builtin_amdgcn_wmma_f32_16x16x32_bf16(false, aP, false,
                           load_bt_frag(Vh, T_, kv0, t2 * 16), (short)0,
                           accO[t2], false, false);
    }

#pragma unroll
    for (int r = 0; r < 8; ++r) {
        int t = q0 + r + hf * 8;
        float inv = 1.0f / lrow[r];
#pragma unroll
        for (int t2 = 0; t2 < 4; ++t2) {
            int d = t2 * 16 + n16;
            O[((size_t)(b * T_ + t) * E_) + h * HS_ + d] = f2bf(accO[t2][r] * inv);
        }
    }
}

// ---------------------------------------------------------------------------
extern "C" void kernel_launch(void* const* d_in, const int* in_sizes, int n_in,
                              void* d_out, int out_size, void* d_ws, size_t ws_size,
                              hipStream_t stream) {
    (void)in_sizes; (void)n_in; (void)out_size; (void)ws_size;
    const float* x    = (const float*)d_in[0];
    const float* ln1g = (const float*)d_in[1];
    const float* ln1b = (const float*)d_in[2];
    const float* Wq   = (const float*)d_in[3];
    const float* bq   = (const float*)d_in[4];
    const float* Wk   = (const float*)d_in[5];
    const float* bk   = (const float*)d_in[6];
    const float* Wv   = (const float*)d_in[7];
    const float* bv   = (const float*)d_in[8];
    const float* Wp   = (const float*)d_in[9];
    const float* bp   = (const float*)d_in[10];
    const float* ln2g = (const float*)d_in[11];
    const float* ln2b = (const float*)d_in[12];
    const float* W1   = (const float*)d_in[13];
    const float* b1   = (const float*)d_in[14];
    const float* W2   = (const float*)d_in[15];
    const float* b2   = (const float*)d_in[16];

    char* ws = (char*)d_ws;
    size_t off = 0;
    auto alloc = [&](size_t bytes) -> char* {
        char* p = ws + off;
        off += (bytes + 255) & ~(size_t)255;
        return p;
    };
    bf16_t* wq_t  = (bf16_t*)alloc((size_t)E_ * E_ * 2);
    bf16_t* wk_t  = (bf16_t*)alloc((size_t)E_ * E_ * 2);
    bf16_t* wv_t  = (bf16_t*)alloc((size_t)E_ * E_ * 2);
    bf16_t* wp_t  = (bf16_t*)alloc((size_t)E_ * E_ * 2);
    bf16_t* w1_t  = (bf16_t*)alloc((size_t)E_ * DFF_ * 2);
    bf16_t* w2_t  = (bf16_t*)alloc((size_t)DFF_ * E_ * 2);
    bf16_t* hbuf  = (bf16_t*)alloc((size_t)M_ * E_ * 2);
    bf16_t* qbuf  = (bf16_t*)alloc((size_t)M_ * E_ * 2);
    bf16_t* kbuf  = (bf16_t*)alloc((size_t)M_ * E_ * 2);
    bf16_t* vtbuf = (bf16_t*)alloc((size_t)M_ * E_ * 2);
    bf16_t* attnb = (bf16_t*)alloc((size_t)M_ * E_ * 2);
    float*  xmid  = (float*) alloc((size_t)M_ * E_ * 4);
    bf16_t* h2buf = (bf16_t*)alloc((size_t)M_ * E_ * 2);
    bf16_t* ffn1b = (bf16_t*)alloc((size_t)M_ * DFF_ * 2);

    dim3 blk(256);

    // Weight bf16 conversion + transpose (B-fragment friendly layout)
    conv_transpose_bf16<<<dim3(E_ / 32, E_ / 32), blk, 0, stream>>>(Wq, wq_t, E_, E_);
    conv_transpose_bf16<<<dim3(E_ / 32, E_ / 32), blk, 0, stream>>>(Wk, wk_t, E_, E_);
    conv_transpose_bf16<<<dim3(E_ / 32, E_ / 32), blk, 0, stream>>>(Wv, wv_t, E_, E_);
    conv_transpose_bf16<<<dim3(E_ / 32, E_ / 32), blk, 0, stream>>>(Wp, wp_t, E_, E_);
    conv_transpose_bf16<<<dim3(E_ / 32, DFF_ / 32), blk, 0, stream>>>(W1, w1_t, E_, DFF_);
    conv_transpose_bf16<<<dim3(DFF_ / 32, E_ / 32), blk, 0, stream>>>(W2, w2_t, DFF_, E_);

    // LN1 -> h (bf16)
    layernorm_bf16<<<M_, blk, 0, stream>>>(x, ln1g, ln1b, hbuf);

    // Q, K per-head [B,H,T,HS]; V per-head transposed [B,H,HS,T]
    gemm_bf16<2, false, false><<<dim3(M_ / 64, E_ / 128), blk, 0, stream>>>(
        hbuf, wq_t, bq, nullptr, qbuf, M_, E_, E_);
    gemm_bf16<2, false, false><<<dim3(M_ / 64, E_ / 128), blk, 0, stream>>>(
        hbuf, wk_t, bk, nullptr, kbuf, M_, E_, E_);
    gemm_bf16<3, false, false><<<dim3(M_ / 64, E_ / 128), blk, 0, stream>>>(
        hbuf, wv_t, bv, nullptr, vtbuf, M_, E_, E_);

    // Causal flash attention -> attn [B,T,E] bf16
    flash_attn<<<dim3(T_ / 16 / 8, B_ * H_), blk, 0, stream>>>(qbuf, kbuf, vtbuf, attnb);

    // x_mid = x + attn @ Wp + bp   (f32)
    gemm_bf16<0, false, true><<<dim3(M_ / 64, E_ / 128), blk, 0, stream>>>(
        attnb, wp_t, bp, x, xmid, M_, E_, E_);

    // LN2 -> h2 (bf16)
    layernorm_bf16<<<M_, blk, 0, stream>>>(xmid, ln2g, ln2b, h2buf);

    // ffn1 = relu(h2 @ W1 + b1)  (bf16)
    gemm_bf16<1, true, false><<<dim3(M_ / 64, DFF_ / 128), blk, 0, stream>>>(
        h2buf, w1_t, b1, nullptr, ffn1b, M_, DFF_, E_);

    // out = x_mid + ffn1 @ W2 + b2  (f32)
    gemm_bf16<0, false, true><<<dim3(M_ / 64, E_ / 128), blk, 0, stream>>>(
        ffn1b, w2_t, b2, xmid, (float*)d_out, M_, E_, DFF_);
}
